// YOLONano_30124900614231
// MI455X (gfx1250) — matrix operand order repaired and checked by reading, not compiled, not run on hardware
//
#include <hip/hip_runtime.h>
#include <hip/hip_bf16.h>
#include <stdint.h>

#define NCLS     20
#define NBOX     6300          // (40^2+20^2+10^2)*3 anchors
#define NSORT    8192          // pow2 pad for bitonic sort
#define NTILE    394           // ceil(NBOX/16) row tiles
#define NSPAD    (NTILE * 16)  // 6304 padded boxes
#define NWORD    197           // NSPAD/32 mask words per row
#define CONF_TH  0.001f
#define NMS_TH   0.5f
#define INPUT_SZ 320.0f

typedef __attribute__((ext_vector_type(16))) _Float16 v16h;
typedef __attribute__((ext_vector_type(8)))  float    v8f;
typedef unsigned long long u64;
typedef unsigned int       u32;

__device__ __forceinline__ float sigmoidf_(float x) { return 1.0f / (1.0f + expf(-x)); }

// Map float -> u32 such that ascending u32 order == DESCENDING float order.
__device__ __forceinline__ u32 desc_key(float x) {
    u32 u   = __float_as_uint(x);
    u32 asc = (u & 0x80000000u) ? ~u : (u | 0x80000000u);
    return ~asc;
}

__device__ __forceinline__ bool iou_gt(float4 a, float areaA, float4 b, float areaB) {
    float xx1 = fmaxf(a.x, b.x), yy1 = fmaxf(a.y, b.y);
    float xx2 = fminf(a.z, b.z), yy2 = fminf(a.w, b.w);
    float inter = fmaxf(1e-28f, xx2 - xx1) * fmaxf(1e-28f, yy2 - yy1);
    float iou   = inter / (areaA + areaB - inter);
    return iou > NMS_TH;
}

// ---------------------------------------------------------------- decode ----
__global__ void decode_kernel(const float* __restrict__ conf,
                              const float* __restrict__ cls,
                              const float* __restrict__ txy,
                              const float* __restrict__ grid,
                              const float* __restrict__ strd,
                              const float* __restrict__ anc,
                              float* __restrict__ boxes,
                              float* __restrict__ scores,
                              int*   __restrict__ clsIdx,
                              u64*   __restrict__ keys) {
    int j = blockIdx.x * blockDim.x + threadIdx.x;
    if (j >= NSORT) return;
    if (j >= NBOX) {                        // padding: sorts after every real key
        keys[j] = 0xFFFFFFFF00000000ull | (u32)j;
        return;
    }
    // score = sigmoid(conf) * softmax(cls); argmax / max over classes
    float cv = sigmoidf_(conf[j]);
    const float* cr = cls + j * NCLS;
    float m = -1e30f;
#pragma unroll
    for (int k = 0; k < NCLS; k++) m = fmaxf(m, cr[k]);
    float e[NCLS]; float s = 0.0f;
#pragma unroll
    for (int k = 0; k < NCLS; k++) { e[k] = expf(cr[k] - m); s += e[k]; }
    float inv = 1.0f / s;
    float best = -1.0f; int bi = 0;
#pragma unroll
    for (int k = 0; k < NCLS; k++) {
        float p = cv * e[k] * inv;
        if (p > best) { best = p; bi = k; }
    }
    // box decode
    int hw = j / 3;
    float tx = txy[j*4+0], ty = txy[j*4+1], tw = txy[j*4+2], th = txy[j*4+3];
    float gx = grid[hw*2+0], gy = grid[hw*2+1];
    float st = strd[j*2+0];
    float aw = anc[j*2+0],  ah = anc[j*2+1];
    float cx = (sigmoidf_(tx) + gx) * st;
    float cy = (sigmoidf_(ty) + gy) * st;
    float bw = expf(tw) * aw;
    float bh = expf(th) * ah;
    float x1 = fminf(fmaxf((cx - bw*0.5f) / INPUT_SZ, 0.0f), 1.0f);
    float y1 = fminf(fmaxf((cy - bh*0.5f) / INPUT_SZ, 0.0f), 1.0f);
    float x2 = fminf(fmaxf((cx + bw*0.5f) / INPUT_SZ, 0.0f), 1.0f);
    float y2 = fminf(fmaxf((cy + bh*0.5f) / INPUT_SZ, 0.0f), 1.0f);
    boxes[j*4+0] = x1; boxes[j*4+1] = y1; boxes[j*4+2] = x2; boxes[j*4+3] = y2;
    scores[j] = best;
    clsIdx[j] = bi;
    float ks = (best >= CONF_TH) ? best : -1.0f;   // invalid -> score -1 (matches ref)
    keys[j] = ((u64)desc_key(ks) << 32) | (u32)j;  // stable: ties -> ascending index
}

// ------------------------------------------------------- bitonic sort -------
__global__ void __launch_bounds__(1024) sort_kernel(const u64* __restrict__ keysIn,
                                                    u64* __restrict__ keysOut) {
    __shared__ u64 sk[NSORT];                       // 64 KB of the 320 KB WGP LDS
    int tid = threadIdx.x;
    for (int i = tid; i < NSORT; i += 1024) sk[i] = keysIn[i];
    __syncthreads();
    for (int k = 2; k <= NSORT; k <<= 1) {
        for (int j = k >> 1; j > 0; j >>= 1) {
            for (int i = tid; i < NSORT; i += 1024) {
                int l = i ^ j;
                if (l > i) {
                    bool asc = ((i & k) == 0);
                    u64 a = sk[i], b = sk[l];
                    if ((a > b) == asc) { sk[i] = b; sk[l] = a; }
                }
            }
            __syncthreads();
        }
    }
    for (int i = tid; i < NSORT; i += 1024) keysOut[i] = sk[i];
}

// ------------------------------------------------------------- gather -------
__global__ void gather_kernel(const u64* __restrict__ skeys,
                              const float* __restrict__ boxes,
                              const float* __restrict__ scores,
                              const int*  __restrict__ clsIdx,
                              float4* __restrict__ sboxes,
                              int* __restrict__ scls,
                              int* __restrict__ svalid,
                              int* __restrict__ sord) {
    int i = blockIdx.x * blockDim.x + threadIdx.x;
    if (i >= NSPAD) return;
    if (i < NBOX) {
        int oj = (int)(skeys[i] & 0xFFFFFFFFull);
        sord[i] = oj;
        const float* bp = boxes + (u64)oj * 4;
        sboxes[i] = make_float4(bp[0], bp[1], bp[2], bp[3]);
        scls[i]   = clsIdx[oj];
        svalid[i] = (scores[oj] >= CONF_TH) ? 1 : 0;
    } else {                                        // pad: class -1 -> all-zero one-hot
        sord[i] = 0; sboxes[i] = make_float4(0, 0, 0, 0);
        scls[i] = -1; svalid[i] = 0;
    }
}

// --------------------------------------- WMMA suppression-matrix kernel -----
// One wave per 16-row x 32-col tile.  Class equality cl[m]==cl[n] is computed
// as OneHot(cl_m) . OneHot(cl_n) over K=32 (20 classes zero-padded) with
// v_wmma_f32_16x16x32_f16; IoU>thresh is VALU; ballot_w32 packs the bits.
__global__ void __launch_bounds__(32) suppress_kernel(const float4* __restrict__ sboxes,
                                                      const int* __restrict__ scls,
                                                      u32* __restrict__ mask) {
    int lane = threadIdx.x;
    int cw   = blockIdx.x;           // column word (32 cols)
    int rt   = blockIdx.y;           // row tile (16 rows)
    int m0   = rt * 16;
    int nb0  = cw * 32;
    int hi   = lane >> 4;            // lane group: 0 => K-low half, 1 => K-high half
    int lo   = lane & 15;

    // A operand (16x32 f16): lane 'lo' owns row M=lo; halves map to K per ISA layout
    int clsA = scls[m0 + lo];
    v16h a;
#pragma unroll
    for (int i = 0; i < 16; i++) {
        int k = i + ((i < 8) ? (hi ? 8 : 0) : (hi ? 16 : 8));
        a[i] = (k == clsA) ? (_Float16)1.0f : (_Float16)0.0f;
    }
    // B operands (32x16 f16): lane 'lo' owns col N=lo; halves map to K in the same
    // order as A (lanes 0-15: K=0..7,16..23; lanes 16-31: K=8..15,24..31)
    int clsB0 = scls[nb0 + lo];
    int clsB1 = scls[nb0 + 16 + lo];
    v16h b0, b1;
#pragma unroll
    for (int i = 0; i < 16; i++) {
        int k = i + ((i < 8) ? (hi ? 8 : 0) : (hi ? 16 : 8));
        b0[i] = (k == clsB0) ? (_Float16)1.0f : (_Float16)0.0f;
        b1[i] = (k == clsB1) ? (_Float16)1.0f : (_Float16)0.0f;
    }
    v8f z = {};
    v8f c0 = __builtin_amdgcn_wmma_f32_16x16x32_f16(false, a, false, b0, (short)0, z, false, false);
    v8f c1 = __builtin_amdgcn_wmma_f32_16x16x32_f16(false, a, false, b1, (short)0, z, false, false);

    // Per-lane column boxes (C/D layout: lane -> N = lane%16)
    float4 cb0 = sboxes[nb0 + lo];
    float4 cb1 = sboxes[nb0 + 16 + lo];
    float areaC0 = (cb0.z - cb0.x) * (cb0.w - cb0.y);
    float areaC1 = (cb1.z - cb1.x) * (cb1.w - cb1.y);

#pragma unroll
    for (int r = 0; r < 8; r++) {
        int m = m0 + r + hi * 8;                     // C/D layout: VGPR r -> M=r (+8 hi)
        float4 rb = sboxes[m];
        float areaR = (rb.z - rb.x) * (rb.w - rb.y);
        bool p0 = iou_gt(rb, areaR, cb0, areaC0) && (c0[r] > 0.5f);
        bool p1 = iou_gt(rb, areaR, cb1, areaC1) && (c1[r] > 0.5f);
        u32 bl0 = __builtin_amdgcn_ballot_w32(p0);   // [15:0]=row m0+r, [31:16]=row m0+r+8
        u32 bl1 = __builtin_amdgcn_ballot_w32(p1);
        u32 wlow  = (bl0 & 0xFFFFu) | (bl1 << 16);          // row m0+r,  cols nb0..nb0+31
        u32 whigh = (bl0 >> 16)     | (bl1 & 0xFFFF0000u);  // row m0+r+8
        if (lane == 0) mask[(u64)(m0 + r)     * NWORD + cw] = wlow;
        if (lane == 1) mask[(u64)(m0 + r + 8) * NWORD + cw] = whigh;
    }
}

// ---------------------------------------------- sequential greedy NMS -------
__global__ void __launch_bounds__(128) nms_scan_kernel(const u32* __restrict__ mask,
                                                       const int* __restrict__ svalid,
                                                       int* __restrict__ keepSorted) {
    __shared__ u32 acc[NWORD];
    __shared__ int kflag;
    int t = threadIdx.x;
    for (int w = t; w < NWORD; w += 128) acc[w] = 0u;
    __syncthreads();
    for (int i = 0; i < NBOX; i++) {
        if (t == 0) {
            u32 word = acc[i >> 5];
            int sup  = (word >> (i & 31)) & 1;
            int k    = svalid[i] && !sup;
            kflag = k;
            keepSorted[i] = k;
        }
        if (t < 64)   // stage next row toward L2 (global_prefetch_b8)
            __builtin_prefetch(&mask[(u64)(i + 1) * NWORD + t * 4], 0, 1);
        __syncthreads();
        if (kflag) {
            const u32* row = mask + (u64)i * NWORD;
            for (int w = t; w < NWORD; w += 128) acc[w] |= row[w];
        }
        __syncthreads();
    }
}

// ----------------------------------------------------------- finalize -------
__global__ void finalize_kernel(const int* __restrict__ sord,
                                const int* __restrict__ keepSorted,
                                const float4* __restrict__ boxes,
                                const float* __restrict__ scores,
                                const int*  __restrict__ clsIdx,
                                float* __restrict__ out) {
    int i = blockIdx.x * blockDim.x + threadIdx.x;
    if (i >= NBOX) return;
    int oj = sord[i];                     // sord is a permutation of 0..NBOX-1
    float4 b = boxes[oj];
    float* o = out + (u64)oj * 7;
    o[0] = b.x; o[1] = b.y; o[2] = b.z; o[3] = b.w;
    o[4] = scores[oj];
    o[5] = (float)clsIdx[oj];
    o[6] = keepSorted[i] ? 1.0f : 0.0f;
}

// ------------------------------------------------------------- launch -------
static inline char* bump(char*& p, size_t bytes) {
    char* r = p;
    p += (bytes + 255) & ~(size_t)255;
    return r;
}

extern "C" void kernel_launch(void* const* d_in, const int* in_sizes, int n_in,
                              void* d_out, int out_size, void* d_ws, size_t ws_size,
                              hipStream_t stream) {
    const float* conf = (const float*)d_in[0];   // (1, N, 1)
    const float* cls  = (const float*)d_in[1];   // (1, N, 20)
    const float* txy  = (const float*)d_in[2];   // (1, HW, 3, 4)
    const float* grid = (const float*)d_in[3];   // (1, HW, 1, 2)
    const float* strd = (const float*)d_in[4];   // (1, HW, 3, 2)
    const float* anc  = (const float*)d_in[5];   // (1, HW, 3, 2)

    char* p = (char*)d_ws;
    float* boxes   = (float*)bump(p, (size_t)NBOX * 4 * sizeof(float));
    float* scores  = (float*)bump(p, (size_t)NBOX * sizeof(float));
    int*   clsIdx  = (int*)  bump(p, (size_t)NBOX * sizeof(int));
    u64*   keys    = (u64*)  bump(p, (size_t)NSORT * sizeof(u64));
    u64*   skeys   = (u64*)  bump(p, (size_t)NSORT * sizeof(u64));
    float4* sboxes = (float4*)bump(p, (size_t)NSPAD * sizeof(float4));
    int*   scls    = (int*)  bump(p, (size_t)NSPAD * sizeof(int));
    int*   svalid  = (int*)  bump(p, (size_t)NSPAD * sizeof(int));
    int*   sord    = (int*)  bump(p, (size_t)NSPAD * sizeof(int));
    int*   keepS   = (int*)  bump(p, (size_t)NSPAD * sizeof(int));
    u32*   mask    = (u32*)  bump(p, (size_t)NSPAD * NWORD * sizeof(u32));

    decode_kernel<<<NSORT / 256, 256, 0, stream>>>(conf, cls, txy, grid, strd, anc,
                                                   boxes, scores, clsIdx, keys);
    sort_kernel<<<1, 1024, 0, stream>>>(keys, skeys);
    gather_kernel<<<(NSPAD + 255) / 256, 256, 0, stream>>>(skeys, boxes, scores, clsIdx,
                                                           sboxes, scls, svalid, sord);
    dim3 sg(NWORD, NTILE);
    suppress_kernel<<<sg, 32, 0, stream>>>(sboxes, scls, mask);
    nms_scan_kernel<<<1, 128, 0, stream>>>(mask, svalid, keepS);
    finalize_kernel<<<(NBOX + 255) / 256, 256, 0, stream>>>(sord, keepS, (const float4*)boxes,
                                                            scores, clsIdx, (float*)d_out);
}
